// PerceiverAttentionCA_31739808317753
// MI455X (gfx1250) — compile-verified, matrix operand order checked by reading, not run on hardware
//
#include <hip/hip_runtime.h>

#define DIMX   3072
#define KVD    2048
#define HEADS  16
#define DHEAD  128
#define INNER  2048
#define NSEQ   2048
#define BATCH  2

typedef __bf16 bf16;
typedef __attribute__((ext_vector_type(16))) __bf16 v16bf;
typedef __attribute__((ext_vector_type(8)))  __bf16 bf16x8;
typedef __attribute__((ext_vector_type(8)))  float  v8f;

#ifndef __has_builtin
#define __has_builtin(x) 0
#endif
#if __has_builtin(__builtin_amdgcn_global_load_async_to_lds_b128)
#define HAVE_ASYNC_LDS 1
#else
#define HAVE_ASYNC_LDS 0
#endif

// Exact pointee type from the builtin's signature: int vector_size(16).
typedef int v4i __attribute__((__vector_size__(16)));
typedef __attribute__((address_space(1))) v4i gv4i;   // global
typedef __attribute__((address_space(3))) v4i lv4i;   // LDS

// Copy 16 bytes global -> LDS. Async (ASYNCcnt-tracked, no VGPR round trip)
// when the gfx1250 builtin is available; plain b128 load/store otherwise.
static __device__ __forceinline__ void copy16_to_lds(bf16* dst, const bf16* src) {
#if HAVE_ASYNC_LDS
    __builtin_amdgcn_global_load_async_to_lds_b128((gv4i*)src, (lv4i*)dst, 0, 0);
#else
    *(bf16x8*)dst = *(const bf16x8*)src;
#endif
}

// Wait for this wave's async copies, then barrier so every wave sees all LDS.
static __device__ __forceinline__ void stage_join() {
#if HAVE_ASYNC_LDS
#if __has_builtin(__builtin_amdgcn_s_wait_asynccnt)
    __builtin_amdgcn_s_wait_asynccnt(0);
#else
    asm volatile("s_wait_asynccnt 0" ::: "memory");
#endif
#endif
    __syncthreads();
}

// Load a 16x32 (MxK) bf16 WMMA A/B fragment for this lane from a row-major
// row pointer (row = lane%16, contiguous K). Per ISA 16-bit layout:
// lane-half h gets K runs [h*8, h*8+8) and [16+h*8, 16+h*8+8).
static __device__ __forceinline__ v16bf load_frag(const bf16* base, int half) {
    v16bf f;
    *(bf16x8*)&f       = *(const bf16x8*)(base + half * 8);
    *((bf16x8*)&f + 1) = *(const bf16x8*)(base + 16 + half * 8);
    return f;
}

static __device__ __forceinline__ v8f v8f_zero() {
    v8f z = {0.f, 0.f, 0.f, 0.f, 0.f, 0.f, 0.f, 0.f};
    return z;
}

// ---------------------------------------------------------------------------
// Weight transpose+convert: out[N][K] bf16 = in[K][N] f32, 64x64 LDS tiles.
// ---------------------------------------------------------------------------
__global__ __launch_bounds__(256) void wt_transpose_kernel(
    const float* __restrict__ in, bf16* __restrict__ out, int K, int N) {
    __shared__ bf16 tile[64][72];
    const int tid = threadIdx.x;
    const int k0 = blockIdx.x * 64, n0 = blockIdx.y * 64;

    for (int i = tid; i < 64 * 16; i += 256) {
        int k = i >> 4, nn = (i & 15) * 4;
        float4 v = *(const float4*)(in + (size_t)(k0 + k) * N + n0 + nn);
        tile[k][nn + 0] = (bf16)v.x;
        tile[k][nn + 1] = (bf16)v.y;
        tile[k][nn + 2] = (bf16)v.z;
        tile[k][nn + 3] = (bf16)v.w;
    }
    __syncthreads();
    for (int i = tid; i < 64 * 8; i += 256) {
        int n = i >> 3, kk = (i & 7) * 8;
        bf16x8 t;
#pragma unroll
        for (int j = 0; j < 8; ++j) t[j] = tile[kk + j][n];
        *(bf16x8*)(out + (size_t)(n0 + n) * K + k0 + kk) = t;
    }
}

// ---------------------------------------------------------------------------
// V transpose: vt[(b,h,d)][n] = kv[(b,n)][INNER + h*DHEAD + d], 64x64 tiles.
// ---------------------------------------------------------------------------
__global__ __launch_bounds__(256) void v_transpose_kernel(
    const bf16* __restrict__ kvb, bf16* __restrict__ vt) {
    __shared__ bf16 tile[64][72];
    const int tid = threadIdx.x;
    const int bh = blockIdx.z;
    const int b = bh / HEADS, h = bh % HEADS;
    const int n0 = blockIdx.x * 64, d0 = blockIdx.y * 64;
    const size_t kvpitch = 2 * INNER;

    for (int i = tid; i < 64 * 8; i += 256) {
        int n = i >> 3, dd = (i & 7) * 8;
        *(bf16x8*)&tile[n][dd] = *(const bf16x8*)(
            kvb + ((size_t)b * NSEQ + n0 + n) * kvpitch + INNER + h * DHEAD +
            d0 + dd);
    }
    __syncthreads();
    for (int i = tid; i < 64 * 8; i += 256) {
        int d = i >> 3, nn = (i & 7) * 8;
        bf16x8 t;
#pragma unroll
        for (int j = 0; j < 8; ++j) t[j] = tile[nn + j][d];
        *(bf16x8*)(vt + ((size_t)bh * DHEAD + d0 + d) * NSEQ + n0 + nn) = t;
    }
}

// ---------------------------------------------------------------------------
// LayerNorm (one block per row) -> bf16
// ---------------------------------------------------------------------------
__global__ __launch_bounds__(256) void ln_to_bf16_kernel(
    const float* __restrict__ x, const float* __restrict__ g,
    const float* __restrict__ b, bf16* __restrict__ out, int cols) {
    const int row = blockIdx.x;
    const int tid = threadIdx.x;
    const float* xr = x + (size_t)row * cols;

    float s = 0.f, ss = 0.f;
    for (int i = tid; i < cols; i += 256) {
        float v = xr[i];
        s += v; ss += v * v;
    }
#pragma unroll
    for (int m = 16; m >= 1; m >>= 1) {
        s  += __shfl_xor(s,  m, 32);
        ss += __shfl_xor(ss, m, 32);
    }
    __shared__ float red[16];
    const int wave = tid >> 5, lane = tid & 31;
    if (lane == 0) { red[wave] = s; red[8 + wave] = ss; }
    __syncthreads();
    if (tid == 0) {
        float ts = 0.f, tss = 0.f;
        for (int i = 0; i < 8; ++i) { ts += red[i]; tss += red[8 + i]; }
        red[0] = ts; red[8] = tss;
    }
    __syncthreads();
    const float inv  = 1.0f / (float)cols;
    const float mu   = red[0] * inv;
    const float var  = red[8] * inv - mu * mu;
    const float rstd = rsqrtf(var + 1e-5f);

    bf16* orow = out + (size_t)row * cols;
    for (int i = tid; i < cols; i += 256)
        orow[i] = (bf16)((xr[i] - mu) * rstd * g[i] + b[i]);
}

// ---------------------------------------------------------------------------
// bf16 GEMM: C(MxN) = A(MxK) @ Bt(NxK)^T, f32 accumulate via WMMA.
// Block tile 128x128x64, 256 threads = 8 waves (4x2), wave tile 32x64.
// Both LDS tiles are contiguous 16B copies (B pre-transposed in global).
// LDS rows padded +8 bf16 so 16 fragment rows land in distinct banks.
// ---------------------------------------------------------------------------
#define GBM 128
#define GBN 128
#define GBK 64
#define GPAD 8

__global__ __launch_bounds__(256) void gemm_bf16_kernel(
    const bf16* __restrict__ A, const bf16* __restrict__ Bt,
    float* __restrict__ Cf, bf16* __restrict__ Cb,
    int M, int N, int K) {
    __shared__ __align__(16) bf16 ldsA[GBM][GBK + GPAD];   // 18 KB
    __shared__ __align__(16) bf16 ldsB[GBN][GBK + GPAD];   // 18 KB, [n][k]

    const int tid  = threadIdx.x;
    const int wave = tid >> 5, lane = tid & 31;
    const int ln16 = lane & 15, half = lane >> 4;
    const int waveM = wave >> 1, waveN = wave & 1;
    const int bm0 = blockIdx.y * GBM;
    const int bn0 = blockIdx.x * GBN;

    v8f acc[2][4];
#pragma unroll
    for (int tm = 0; tm < 2; ++tm)
#pragma unroll
        for (int tn = 0; tn < 4; ++tn) acc[tm][tn] = v8f_zero();

    for (int k0 = 0; k0 < K; k0 += GBK) {
        __syncthreads();
        // Stage A tile: 128 x 64 contiguous 16B chunks.
        for (int i = tid; i < GBM * (GBK / 8); i += 256) {
            int m = i >> 3, kk = (i & 7) * 8;
            const bf16* src = A + (size_t)(bm0 + m) * K + k0 + kk;
            copy16_to_lds(&ldsA[m][kk], src);
            __builtin_prefetch(src + GBK, 0, 0);  // next k-tile -> L2
        }
        // Stage B tile (already [n][k] in global): contiguous copies.
        for (int i = tid; i < GBN * (GBK / 8); i += 256) {
            int n = i >> 3, kk = (i & 7) * 8;
            const bf16* src = Bt + (size_t)(bn0 + n) * K + k0 + kk;
            copy16_to_lds(&ldsB[n][kk], src);
            __builtin_prefetch(src + GBK, 0, 0);
        }
        stage_join();

#pragma unroll
        for (int c = 0; c < GBK / 32; ++c) {
            v16bf af[2], bfr[4];
#pragma unroll
            for (int tm = 0; tm < 2; ++tm)
                af[tm] = load_frag(&ldsA[waveM * 32 + tm * 16 + ln16][c * 32], half);
#pragma unroll
            for (int tn = 0; tn < 4; ++tn)
                bfr[tn] = load_frag(&ldsB[waveN * 64 + tn * 16 + ln16][c * 32], half);
#pragma unroll
            for (int tm = 0; tm < 2; ++tm)
#pragma unroll
                for (int tn = 0; tn < 4; ++tn)
                    acc[tm][tn] = __builtin_amdgcn_wmma_f32_16x16x32_bf16(
                        false, af[tm], false, bfr[tn], (short)0, acc[tm][tn],
                        false, false);
        }
    }

    // C layout: lane n = lane%16, VGPR r -> m = r + 8*(lane/16).
#pragma unroll
    for (int tm = 0; tm < 2; ++tm)
#pragma unroll
        for (int tn = 0; tn < 4; ++tn)
#pragma unroll
            for (int r = 0; r < 8; ++r) {
                int row = bm0 + waveM * 32 + tm * 16 + r + 8 * half;
                int col = bn0 + waveN * 64 + tn * 16 + ln16;
                float v = acc[tm][tn][r];
                if (Cf) Cf[(size_t)row * N + col] = v;
                else    Cb[(size_t)row * N + col] = (bf16)v;
            }
}

// ---------------------------------------------------------------------------
// Flash attention, one block per (128 q rows, head, batch). 8 waves, each
// owns 16 q rows. Q fragments register-resident; K tile + pre-transposed V
// tile staged with async contiguous copies; P re-laid out C->A through a
// per-wave LDS buffer; online softmax in fp32.
// ---------------------------------------------------------------------------
__global__ __launch_bounds__(256) void attn_kernel(
    const bf16* __restrict__ qbuf,   // (B*NSEQ, INNER)
    const bf16* __restrict__ kvbuf,  // (B*NSEQ, 2*INNER)  [k | v]
    const bf16* __restrict__ vt,     // (B*H*DHEAD, NSEQ)  v transposed
    bf16* __restrict__ obuf) {       // (B*NSEQ, INNER)
    __shared__ __align__(16) bf16 ldsK[64][DHEAD + 8];   // ~17 KB
    __shared__ __align__(16) bf16 ldsVt[DHEAD][64 + 8];  // 18 KB
    __shared__ __align__(16) bf16 sbuf[8][16][64 + 8];   // 18 KB

    const int tid  = threadIdx.x;
    const int wave = tid >> 5, lane = tid & 31;
    const int ln16 = lane & 15, half = lane >> 4;
    const int head = blockIdx.y, batch = blockIdx.z;
    const size_t qrow0   = (size_t)batch * NSEQ + (size_t)blockIdx.x * 128;
    const size_t krow0   = (size_t)batch * NSEQ;
    const size_t kvpitch = 2 * INNER;
    const bf16* vtbase = vt + ((size_t)batch * HEADS + head) * DHEAD * NSEQ;

    // Q fragments (16 rows x 128 d = 4 fragments), direct from global.
    v16bf qa[4];
    {
        const bf16* qr =
            qbuf + (qrow0 + wave * 16 + ln16) * (size_t)INNER + head * DHEAD;
#pragma unroll
        for (int c = 0; c < 4; ++c) qa[c] = load_frag(qr + c * 32, half);
    }

    v8f o[8];
#pragma unroll
    for (int j = 0; j < 8; ++j) o[j] = v8f_zero();
    float rmax[8], rsum[8];
#pragma unroll
    for (int r = 0; r < 8; ++r) { rmax[r] = -3.0e38f; rsum[r] = 0.f; }

    const float sscale = 0.088388347648318447f;  // 1/sqrt(128) (q and k scales)

    for (int kb = 0; kb < NSEQ; kb += 64) {
        __syncthreads();
        // Stage K tile [key][d] (natural layout) - contiguous 16B copies.
        for (int i = tid; i < 64 * (DHEAD / 8); i += 256) {
            int key = i >> 4, dv = (i & 15) * 8;
            copy16_to_lds(&ldsK[key][dv],
                          kvbuf + (krow0 + kb + key) * kvpitch + head * DHEAD + dv);
        }
        // Stage V^T tile [d][key] from pre-transposed global - contiguous.
        for (int i = tid; i < DHEAD * 8; i += 256) {
            int d = i >> 3, kf = (i & 7) * 8;
            copy16_to_lds(&ldsVt[d][kf], vtbase + (size_t)d * NSEQ + kb + kf);
        }
        stage_join();

        // S = Q @ K^T : 4 key-tiles x 4 d-chunks.
        v8f s[4];
#pragma unroll
        for (int t = 0; t < 4; ++t) s[t] = v8f_zero();
#pragma unroll
        for (int t = 0; t < 4; ++t)
#pragma unroll
            for (int c = 0; c < 4; ++c) {
                v16bf kf = load_frag(&ldsK[t * 16 + ln16][c * 32], half);
                s[t] = __builtin_amdgcn_wmma_f32_16x16x32_bf16(
                    false, qa[c], false, kf, (short)0, s[t], false, false);
            }

        // Scale + per-row block max (reduce over t, then 16-lane N-group).
        float cmax[8];
#pragma unroll
        for (int r = 0; r < 8; ++r) cmax[r] = -3.0e38f;
#pragma unroll
        for (int t = 0; t < 4; ++t)
#pragma unroll
            for (int r = 0; r < 8; ++r) {
                s[t][r] *= sscale;
                cmax[r] = fmaxf(cmax[r], s[t][r]);
            }
#pragma unroll
        for (int m = 8; m >= 1; m >>= 1)
#pragma unroll
            for (int r = 0; r < 8; ++r)
                cmax[r] = fmaxf(cmax[r], __shfl_xor(cmax[r], m, 32));

        // Online softmax rescale.
        float corr[8];
#pragma unroll
        for (int r = 0; r < 8; ++r) {
            float nm = fmaxf(rmax[r], cmax[r]);
            corr[r] = __expf(rmax[r] - nm);
            rmax[r] = nm;
            rsum[r] *= corr[r];
        }
#pragma unroll
        for (int j = 0; j < 8; ++j)
#pragma unroll
            for (int r = 0; r < 8; ++r) o[j][r] *= corr[r];

        // P = exp(S - max), bf16 via per-wave LDS buffer [m][key].
        float psum[8];
#pragma unroll
        for (int r = 0; r < 8; ++r) psum[r] = 0.f;
#pragma unroll
        for (int t = 0; t < 4; ++t)
#pragma unroll
            for (int r = 0; r < 8; ++r) {
                float p = __expf(s[t][r] - rmax[r]);
                psum[r] += p;
                sbuf[wave][r + 8 * half][t * 16 + ln16] = (bf16)p;
            }
#pragma unroll
        for (int m = 8; m >= 1; m >>= 1)
#pragma unroll
            for (int r = 0; r < 8; ++r) psum[r] += __shfl_xor(psum[r], m, 32);
#pragma unroll
        for (int r = 0; r < 8; ++r) rsum[r] += psum[r];

        // O += P @ V : 2 key-chunks x 8 d-tiles (same-wave LDS RAW; compiler
        // inserts s_wait_dscnt).
#pragma unroll
        for (int c = 0; c < 2; ++c) {
            v16bf pf = load_frag(&sbuf[wave][ln16][c * 32], half);
#pragma unroll
            for (int j = 0; j < 8; ++j) {
                v16bf vf = load_frag(&ldsVt[j * 16 + ln16][c * 32], half);
                o[j] = __builtin_amdgcn_wmma_f32_16x16x32_bf16(
                    false, pf, false, vf, (short)0, o[j], false, false);
            }
        }
    }

    // Normalize and store (b, n2, h, d) flattened = (B*NSEQ, INNER).
#pragma unroll
    for (int j = 0; j < 8; ++j)
#pragma unroll
        for (int r = 0; r < 8; ++r) {
            size_t row = qrow0 + wave * 16 + r + 8 * half;
            float v = o[j][r] / rsum[r];
            obuf[row * INNER + head * DHEAD + j * 16 + ln16] = (bf16)v;
        }
}

// ---------------------------------------------------------------------------
// Host launcher
// ---------------------------------------------------------------------------
extern "C" void kernel_launch(void* const* d_in, const int* in_sizes, int n_in,
                              void* d_out, int out_size, void* d_ws,
                              size_t ws_size, hipStream_t stream) {
    const float* x     = (const float*)d_in[0];
    const float* lat   = (const float*)d_in[1];
    const float* w_q   = (const float*)d_in[2];
    const float* w_kv  = (const float*)d_in[3];
    const float* w_out = (const float*)d_in[4];
    const float* ln1_g = (const float*)d_in[5];
    const float* ln1_b = (const float*)d_in[6];
    const float* ln2_g = (const float*)d_in[7];
    const float* ln2_b = (const float*)d_in[8];
    float* out = (float*)d_out;

    char* ws = (char*)d_ws;
    size_t off = 0;
    auto alloc = [&](size_t bytes) -> void* {
        void* p = ws + off;
        off += (bytes + 255) & ~(size_t)255;
        return p;
    };
    const size_t rows = (size_t)BATCH * NSEQ;  // 4096
    bf16* xn    = (bf16*)alloc(rows * KVD * 2);
    bf16* lnl   = (bf16*)alloc(rows * DIMX * 2);
    bf16* wq_t  = (bf16*)alloc((size_t)DIMX * INNER * 2);       // [INNER][DIMX]
    bf16* wkv_t = (bf16*)alloc((size_t)KVD * 2 * INNER * 2);    // [2*INNER][KVD]
    bf16* wo_t  = (bf16*)alloc((size_t)INNER * DIMX * 2);       // [DIMX][INNER]
    bf16* qb    = (bf16*)alloc(rows * INNER * 2);
    bf16* kvb   = (bf16*)alloc(rows * 2 * INNER * 2);
    bf16* att   = (bf16*)alloc(rows * INNER * 2);
    bf16* vtb   = (bf16*)alloc((size_t)BATCH * HEADS * DHEAD * NSEQ * 2);

    dim3 blk(256);

    // Weight transpose+convert f32[K][N] -> bf16[N][K].
    wt_transpose_kernel<<<dim3(DIMX / 64, INNER / 64), blk, 0, stream>>>(
        w_q, wq_t, DIMX, INNER);
    wt_transpose_kernel<<<dim3(KVD / 64, 2 * INNER / 64), blk, 0, stream>>>(
        w_kv, wkv_t, KVD, 2 * INNER);
    wt_transpose_kernel<<<dim3(INNER / 64, DIMX / 64), blk, 0, stream>>>(
        w_out, wo_t, INNER, DIMX);

    // LayerNorms -> bf16.
    ln_to_bf16_kernel<<<dim3((int)rows), blk, 0, stream>>>(x, ln1_g, ln1_b, xn,
                                                           KVD);
    ln_to_bf16_kernel<<<dim3((int)rows), blk, 0, stream>>>(lat, ln2_g, ln2_b,
                                                           lnl, DIMX);

    // q = ln(latents) @ w_q  : (4096 x 2048), K = 3072.
    gemm_bf16_kernel<<<dim3(INNER / GBN, (int)rows / GBM), blk, 0, stream>>>(
        lnl, wq_t, nullptr, qb, (int)rows, INNER, DIMX);

    // kv = ln(x) @ w_kv      : (4096 x 4096), K = 2048.
    gemm_bf16_kernel<<<dim3(2 * INNER / GBN, (int)rows / GBM), blk, 0, stream>>>(
        xn, wkv_t, nullptr, kvb, (int)rows, 2 * INNER, KVD);

    // V transpose for the PV matmul.
    v_transpose_kernel<<<dim3(NSEQ / 64, DHEAD / 64, BATCH * HEADS), blk, 0,
                         stream>>>(kvb, vtb);

    // Attention.
    attn_kernel<<<dim3(NSEQ / 128, HEADS, BATCH), blk, 0, stream>>>(qb, kvb,
                                                                    vtb, att);

    // out = attn @ w_out     : (4096 x 3072), K = 2048, fp32 output.
    gemm_bf16_kernel<<<dim3(DIMX / GBN, (int)rows / GBM), blk, 0, stream>>>(
        att, wo_t, out, nullptr, (int)rows, DIMX, INNER);
}
